// PointNetSetAbstraction_59717225284067
// MI455X (gfx1250) — matrix operand order repaired and checked by reading, not compiled, not run on hardware
//
#include <hip/hip_runtime.h>

typedef __attribute__((ext_vector_type(16))) _Float16 v16h;
typedef __attribute__((ext_vector_type(8)))  float    v8f;

#define B_    16
#define N_    4096
#define S_    1024
#define K_    32
#define D_    64
#define NPTS  (B_ * S_ * K_)     // 524288
#define C0A   67                 // 3 + D
#define C0P   96                 // padded to multiple of 32
#define EPSBN 1e-5f
#define NEWXYZ (B_ * S_ * 3)     // 49152 floats at start of d_out

// ---------------------------------------------------------------------------
// Wave32 argmax/argmin butterfly (tie -> lower index), no barriers.
// ---------------------------------------------------------------------------
__device__ __forceinline__ void wave_argmax(float& v, int& i) {
#pragma unroll
  for (int m = 16; m > 0; m >>= 1) {
    float ov = __shfl_xor(v, m, 32);
    int   oi = __shfl_xor(i, m, 32);
    if (ov > v || (ov == v && oi < i)) { v = ov; i = oi; }
  }
}
__device__ __forceinline__ void wave_argmin(float& v, int& i) {
#pragma unroll
  for (int m = 16; m > 0; m >>= 1) {
    float ov = __shfl_xor(v, m, 32);
    int   oi = __shfl_xor(i, m, 32);
    if (ov < v || (ov == v && oi < i)) { v = ov; i = oi; }
  }
}

// ---------------------------------------------------------------------------
// 1) Farthest point sampling: one block per batch; 1024 dependent rounds.
//    Coords + running distances live in VGPRs; shuffle reduce + tiny LDS hop.
// ---------------------------------------------------------------------------
__global__ __launch_bounds__(256) void fps_kernel(const float* __restrict__ xyz,
                                                  int* __restrict__ fidx,
                                                  float* __restrict__ newxyz) {
  const int b = blockIdx.x;
  const int t = threadIdx.x;
  const float* base = xyz + (size_t)b * N_ * 3;

  float px[16], py[16], pz[16], dst[16];
#pragma unroll
  for (int j = 0; j < 16; ++j) {
    int p = t * 16 + j;
    px[j] = base[p * 3 + 0];
    py[j] = base[p * 3 + 1];
    pz[j] = base[p * 3 + 2];
    dst[j] = 1e10f;
  }

  __shared__ float wv[8];
  __shared__ int   wi[8];
  __shared__ int   s_far;
  const int wave = t >> 5;

  int far = 0;
  for (int it = 0; it < S_; ++it) {
    if (t == 0) {
      fidx[b * S_ + it] = far;
      float* o = newxyz + (size_t)(b * S_ + it) * 3;
      o[0] = base[far * 3 + 0];
      o[1] = base[far * 3 + 1];
      o[2] = base[far * 3 + 2];
    }
    const float cx = base[far * 3 + 0];
    const float cy = base[far * 3 + 1];
    const float cz = base[far * 3 + 2];

    float bv = -1.0f;
    int   bi = 0x7fffffff;
#pragma unroll
    for (int j = 0; j < 16; ++j) {
      float dx = px[j] - cx, dy = py[j] - cy, dz = pz[j] - cz;
      float d = dx * dx + dy * dy + dz * dz;
      d = fminf(dst[j], d);
      dst[j] = d;
      int gi = t * 16 + j;
      if (d > bv || (d == bv && gi < bi)) { bv = d; bi = gi; }
    }
    wave_argmax(bv, bi);
    if ((t & 31) == 0) { wv[wave] = bv; wi[wave] = bi; }
    __syncthreads();
    if (t == 0) {
      float fv = wv[0]; int fi = wi[0];
#pragma unroll
      for (int w = 1; w < 8; ++w)
        if (wv[w] > fv || (wv[w] == fv && wi[w] < fi)) { fv = wv[w]; fi = wi[w]; }
      s_far = fi;
    }
    __syncthreads();
    far = s_far;
  }
}

// ---------------------------------------------------------------------------
// 2) kNN: one block per centroid; distances in LDS, 32 masked argmin rounds.
// ---------------------------------------------------------------------------
__global__ __launch_bounds__(128) void knn_kernel(const float* __restrict__ xyz,
                                                  const float* __restrict__ newxyz,
                                                  int* __restrict__ idxOut) {
  const int bs = blockIdx.x;     // b*S + s
  const int b  = bs >> 10;
  const int t  = threadIdx.x;
  const float* base = xyz + (size_t)b * N_ * 3;
  const float cx = newxyz[bs * 3 + 0];
  const float cy = newxyz[bs * 3 + 1];
  const float cz = newxyz[bs * 3 + 2];

  __shared__ float dist[N_];
  __shared__ float wv[4];
  __shared__ int   wi[4];
  const int wave = t >> 5;

  for (int p = t; p < N_; p += 128) {
    float dx = base[p * 3 + 0] - cx;
    float dy = base[p * 3 + 1] - cy;
    float dz = base[p * 3 + 2] - cz;
    dist[p] = dx * dx + dy * dy + dz * dz;
  }
  __syncthreads();

  for (int kq = 0; kq < K_; ++kq) {
    float bv = 3.0e38f;
    int   bi = 0x7fffffff;
    for (int p = t; p < N_; p += 128) {
      float d = dist[p];
      if (d < bv || (d == bv && p < bi)) { bv = d; bi = p; }
    }
    wave_argmin(bv, bi);
    if ((t & 31) == 0) { wv[wave] = bv; wi[wave] = bi; }
    __syncthreads();
    if (t == 0) {
      float fv = wv[0]; int fi = wi[0];
#pragma unroll
      for (int w = 1; w < 4; ++w)
        if (wv[w] < fv || (wv[w] == fv && wi[w] < fi)) { fv = wv[w]; fi = wi[w]; }
      idxOut[(size_t)bs * K_ + kq] = fi;
      dist[fi] = 3.0e38f;
    }
    __syncthreads();
  }
}

// ---------------------------------------------------------------------------
// 3) Gather + feature build (vectorized 16B stores): X0[pt][0..2]=xyz-c,
//    [3..66]=feats, [67..95]=0.  Point-major f16 feeds WMMA B-fragments.
// ---------------------------------------------------------------------------
__global__ __launch_bounds__(256) void gather_kernel(const float* __restrict__ xyz,
                                                     const float* __restrict__ pts,
                                                     const float* __restrict__ newxyz,
                                                     const int* __restrict__ idx,
                                                     _Float16* __restrict__ X0) {
  const int t  = blockIdx.x * 256 + threadIdx.x;   // 0..NPTS-1
  const int bs = t >> 5;                           // /K_
  const int b  = bs >> 10;
  const int n  = idx[t];
  const float* p = xyz + ((size_t)b * N_ + n) * 3;
  const float* c = newxyz + (size_t)bs * 3;

  union { _Float16 h[C0P]; uint4 v[C0P / 8]; } o;
  o.h[0] = (_Float16)(p[0] - c[0]);
  o.h[1] = (_Float16)(p[1] - c[1]);
  o.h[2] = (_Float16)(p[2] - c[2]);
  const float4* f = (const float4*)(pts + ((size_t)b * N_ + n) * D_);
#pragma unroll
  for (int j = 0; j < D_ / 4; ++j) {
    float4 x = f[j];
    o.h[3 + j * 4 + 0] = (_Float16)x.x;
    o.h[3 + j * 4 + 1] = (_Float16)x.y;
    o.h[3 + j * 4 + 2] = (_Float16)x.z;
    o.h[3 + j * 4 + 3] = (_Float16)x.w;
  }
#pragma unroll
  for (int j = C0A; j < C0P; ++j) o.h[j] = (_Float16)0.0f;

  uint4* dstv = (uint4*)(X0 + (size_t)t * C0P);
#pragma unroll
  for (int j = 0; j < C0P / 8; ++j) dstv[j] = o.v[j];
}

// ---------------------------------------------------------------------------
// 4a) One-time W convert: f32 [COUT][CINA] -> padded f16 [COUT][CINP]
// ---------------------------------------------------------------------------
__global__ void wconv_kernel(const float* __restrict__ Wg, _Float16* __restrict__ Wc,
                             int CA, int CP, int CO) {
  for (int e = blockIdx.x * 256 + threadIdx.x; e < CO * CP; e += gridDim.x * 256) {
    int r = e / CP, c = e % CP;
    Wc[e] = (c < CA) ? (_Float16)Wg[r * CA + c] : (_Float16)0.0f;
  }
}

// ---------------------------------------------------------------------------
// 4b) WMMA GEMM: Y[pt][COUT] = W[COUT][CINP] * X[pt][CINP]  (f16 in, f32 acc)
//     256 thr / 8 waves, 128-point tile. W + X staged into LDS via the CDNA5
//     async-LDS path (global_load_async_to_lds_b128 + s_wait_asynccnt).
//     A fragment fixed per wave (mt = wave % numM), reused across N-tiles.
// ---------------------------------------------------------------------------
template <int CINP, int COUT>
__global__ __launch_bounds__(256) void gemm_wmma_kernel(const _Float16* __restrict__ Wc,
                                                        const _Float16* __restrict__ Xg,
                                                        _Float16* __restrict__ Yg) {
  __shared__ _Float16 Wl[COUT * CINP];
  __shared__ _Float16 Xl[128 * CINP];
  const int t = threadIdx.x;
  const long ptBase = (long)blockIdx.x * 128;

  {  // async stage: W tile + X tile, 16B per lane per issue
    constexpr int NW = COUT * CINP * 2 / 16;
    const unsigned wbase = (unsigned)(size_t)(void*)&Wl[0];
    for (int e = t; e < NW; e += 256) {
      unsigned long long g = (unsigned long long)(size_t)((const char*)Wc + (size_t)e * 16);
      unsigned l = wbase + e * 16;
      asm volatile("global_load_async_to_lds_b128 %0, %1, off" :: "v"(l), "v"(g) : "memory");
    }
    constexpr int NX = 128 * CINP * 2 / 16;
    const unsigned xbase = (unsigned)(size_t)(void*)&Xl[0];
    const char* src = (const char*)(Xg + ptBase * CINP);
    for (int e = t; e < NX; e += 256) {
      unsigned long long g = (unsigned long long)(size_t)(src + (size_t)e * 16);
      unsigned l = xbase + e * 16;
      asm volatile("global_load_async_to_lds_b128 %0, %1, off" :: "v"(l), "v"(g) : "memory");
    }
    asm volatile("s_wait_asynccnt 0x0" ::: "memory");
  }
  __syncthreads();

  const int lane = t & 31;
  const int wave = t >> 5;
  const int l15  = lane & 15;
  const int lh   = lane >> 4;
  constexpr int numM   = COUT / 16;
  constexpr int KS     = CINP / 32;
  constexpr int ntStep = 8 / numM;          // 2 (COUT=64) or 1 (COUT=128)
  const int mt  = wave % numM;
  const int nt0 = wave / numM;

  // A fragments for this wave's M-row, all K-steps (CDNA5 16-bit A 16x32 layout)
  union { v16h h; unsigned u[8]; } Af[KS];
  {
    const unsigned* wrow = (const unsigned*)(Wl + (mt * 16 + l15) * CINP);
#pragma unroll
    for (int ks = 0; ks < KS; ++ks) {
      const int ka = (ks * 32 + lh * 8) >> 1;   // K = ks*32 + lh*8 + {0..7, 16..23}
#pragma unroll
      for (int p = 0; p < 4; ++p) {
        Af[ks].u[p]     = wrow[ka + p];
        Af[ks].u[4 + p] = wrow[ka + 8 + p];
      }
    }
  }

  for (int nt = nt0; nt < 8; nt += ntStep) {
    v8f acc = {};
    const unsigned* xrow = (const unsigned*)(Xl + (nt * 16 + l15) * CINP);
#pragma unroll
    for (int ks = 0; ks < KS; ++ks) {
      union { v16h h; unsigned u[8]; } Bf;
      const int kb = (ks * 32 + lh * 8) >> 1;   // K = ks*32 + lh*8 + {0..7, 16..23}
#pragma unroll
      for (int p = 0; p < 4; ++p) { Bf.u[p] = xrow[kb + p]; Bf.u[4 + p] = xrow[kb + 8 + p]; }
      acc = __builtin_amdgcn_wmma_f32_16x16x32_f16(false, Af[ks].h, false, Bf.h,
                                                   (short)0, acc, false, false);
    }
    union { _Float16 h[8]; uint4 v; } outv;
#pragma unroll
    for (int r = 0; r < 8; ++r) outv.h[r] = (_Float16)acc[r];
    const long pt = ptBase + nt * 16 + l15;     // C/D: col=lane&15, rows lh*8..+7
    *reinterpret_cast<uint4*>(Yg + pt * COUT + mt * 16 + lh * 8) = outv.v;
  }
}

// ---------------------------------------------------------------------------
// 5) BatchNorm stats: vectorized 8-halves/thread; fixed 8-channel group per
//    thread; LDS reduce then per-channel global f32 atomics.
// ---------------------------------------------------------------------------
__global__ void zero_stats_kernel(float* stats, int n) {
  int t = blockIdx.x * blockDim.x + threadIdx.x;
  if (t < n) stats[t] = 0.0f;
}

template <int COUT>
__global__ __launch_bounds__(256) void stats_kernel(const _Float16* __restrict__ Y,
                                                    float* __restrict__ sum,
                                                    float* __restrict__ sumsq) {
  constexpr int R = COUT / 8;
  const long nv     = (long)NPTS * COUT / 8;
  const long i0     = (long)blockIdx.x * 256 + threadIdx.x;
  const long stride = (long)gridDim.x * 256;   // *8 is multiple of COUT
  float s[8] = {}, q[8] = {};
  for (long g = i0; g < nv; g += stride) {
    union { _Float16 h[8]; uint4 v; } y;
    y.v = *(const uint4*)(Y + g * 8);
#pragma unroll
    for (int j = 0; j < 8; ++j) { float v = (float)y.h[j]; s[j] += v; q[j] += v * v; }
  }
  __shared__ float ss[2048], sq[2048];
  const int t = threadIdx.x;
#pragma unroll
  for (int j = 0; j < 8; ++j) { ss[t * 8 + j] = s[j]; sq[t * 8 + j] = q[j]; }
  __syncthreads();
  if (t < COUT) {                       // channel t held by threads t2 % R == t>>3, slot t&7
    const int r = t >> 3, j = t & 7;
    float a = 0.0f, b2 = 0.0f;
    for (int t2 = r; t2 < 256; t2 += R) { a += ss[t2 * 8 + j]; b2 += sq[t2 * 8 + j]; }
    atomicAdd(&sum[t], a);
    atomicAdd(&sumsq[t], b2);
  }
}

__global__ void finalize_stats_kernel(const float* sum, const float* sumsq,
                                      const float* g, const float* bt,
                                      float* Ac, float* Bc, int cout, float invN) {
  int t = threadIdx.x;
  if (t < cout) {
    float m  = sum[t] * invN;
    float v  = sumsq[t] * invN - m * m;   // biased var (torch BN training)
    float is = rsqrtf(v + EPSBN);
    float a  = g[t] * is;
    Ac[t] = a;
    Bc[t] = bt[t] - m * a;
  }
}

// ---------------------------------------------------------------------------
// 6) BN + ReLU (layers 1,2), vectorized: x = relu(y*A[c] + B[c])
// ---------------------------------------------------------------------------
template <int COUT>
__global__ __launch_bounds__(256) void bnrelu_kernel(const _Float16* __restrict__ Y,
                                                     const float* __restrict__ Ac,
                                                     const float* __restrict__ Bc,
                                                     _Float16* __restrict__ X) {
  const long e0 = ((long)blockIdx.x * 256 + threadIdx.x) * 8;
  const int  c0 = (int)(e0 & (COUT - 1));
  union { _Float16 h[8]; uint4 v; } y, x;
  y.v = *(const uint4*)(Y + e0);
#pragma unroll
  for (int j = 0; j < 8; ++j)
    x.h[j] = (_Float16)fmaxf(0.0f, fmaf((float)y.h[j], Ac[c0 + j], Bc[c0 + j]));
  *(uint4*)(X + e0) = x.v;
}

// ---------------------------------------------------------------------------
// 7) Layer-3 epilogue: BN + ReLU + max over K, 8 channels per thread,
//    16B neighbor loads -> d_out [B,128,S]
// ---------------------------------------------------------------------------
__global__ __launch_bounds__(256) void bnmax_out_kernel(const _Float16* __restrict__ Y,
                                                        const float* __restrict__ Ac,
                                                        const float* __restrict__ Bc,
                                                        float* __restrict__ out) {
  const int t  = blockIdx.x * 256 + threadIdx.x;  // B*S*16 threads
  const int m0 = (t & 15) * 8;
  const int bs = t >> 4;
  const int b  = bs >> 10;
  const int s  = bs & 1023;
  float a[8], bb[8], mx[8];
#pragma unroll
  for (int j = 0; j < 8; ++j) { a[j] = Ac[m0 + j]; bb[j] = Bc[m0 + j]; mx[j] = 0.0f; }
  const _Float16* y = Y + (size_t)bs * K_ * 128 + m0;
#pragma unroll 4
  for (int k = 0; k < K_; ++k) {
    union { _Float16 h[8]; uint4 v; } yv;
    yv.v = *(const uint4*)(y + (size_t)k * 128);
#pragma unroll
    for (int j = 0; j < 8; ++j)
      mx[j] = fmaxf(mx[j], fmaxf(0.0f, fmaf((float)yv.h[j], a[j], bb[j])));
  }
  float* o = out + (size_t)b * (128 * S_) + (size_t)m0 * S_ + s;
#pragma unroll
  for (int j = 0; j < 8; ++j) o[j * S_] = mx[j];
}

// ---------------------------------------------------------------------------
// Host launcher. Workspace layout (bytes):
//   [0)        fidx   int[16384]
//   [65536)    idx    int[524288]
//   [2162688)  sum[128] + sumsq[128]
//   [2163712)  Ac[128] + Bc[128]
//   [2164736)  Wc     f16[<=128*96]          (32 KB reserved)
//   [2197504)  Xbuf   f16[128*NPTS]          (134 MB)
//   [+134MB)   Ybuf   f16[128*NPTS]          (134 MB)
// ---------------------------------------------------------------------------
extern "C" void kernel_launch(void* const* d_in, const int* in_sizes, int n_in,
                              void* d_out, int out_size, void* d_ws, size_t ws_size,
                              hipStream_t stream) {
  const float* xyz = (const float*)d_in[0];
  const float* pts = (const float*)d_in[1];
  const float* W1  = (const float*)d_in[2];
  // d_in[3,7,11] = biases: absorbed by BN mean subtraction -> unused
  const float* g1  = (const float*)d_in[4];
  const float* bt1 = (const float*)d_in[5];
  const float* W2  = (const float*)d_in[6];
  const float* g2  = (const float*)d_in[8];
  const float* bt2 = (const float*)d_in[9];
  const float* W3  = (const float*)d_in[10];
  const float* g3  = (const float*)d_in[12];
  const float* bt3 = (const float*)d_in[13];

  float* outp    = (float*)d_out;
  float* newxyz  = outp;            // [B,S,3]
  float* outFeat = outp + NEWXYZ;   // [B,128,S]

  char* ws = (char*)d_ws;
  int*      fidx  = (int*)ws;
  int*      idx   = (int*)(ws + 65536);
  float*    sum   = (float*)(ws + 2162688);
  float*    sumsq = sum + 128;
  float*    Ac    = (float*)(ws + 2163712);
  float*    Bc    = Ac + 128;
  _Float16* Wc    = (_Float16*)(ws + 2164736);
  _Float16* Xb    = (_Float16*)(ws + 2197504);
  _Float16* Yb    = (_Float16*)(ws + 2197504 + (size_t)134217728);

  const float invN = 1.0f / (float)NPTS;

  // --- sample_and_group ---
  fps_kernel<<<B_, 256, 0, stream>>>(xyz, fidx, newxyz);
  knn_kernel<<<B_ * S_, 128, 0, stream>>>(xyz, newxyz, idx);
  gather_kernel<<<NPTS / 256, 256, 0, stream>>>(xyz, pts, newxyz, idx, Xb);

  // --- layer 1: 67 -> 64 ---
  wconv_kernel<<<8, 256, 0, stream>>>(W1, Wc, C0A, C0P, 64);
  zero_stats_kernel<<<1, 256, 0, stream>>>(sum, 256);
  gemm_wmma_kernel<C0P, 64><<<NPTS / 128, 256, 0, stream>>>(Wc, Xb, Yb);
  stats_kernel<64><<<512, 256, 0, stream>>>(Yb, sum, sumsq);
  finalize_stats_kernel<<<1, 128, 0, stream>>>(sum, sumsq, g1, bt1, Ac, Bc, 64, invN);
  bnrelu_kernel<64><<<(int)(((long)NPTS * 64 / 8) / 256), 256, 0, stream>>>(Yb, Ac, Bc, Xb);

  // --- layer 2: 64 -> 64 ---
  wconv_kernel<<<8, 256, 0, stream>>>(W2, Wc, 64, 64, 64);
  zero_stats_kernel<<<1, 256, 0, stream>>>(sum, 256);
  gemm_wmma_kernel<64, 64><<<NPTS / 128, 256, 0, stream>>>(Wc, Xb, Yb);
  stats_kernel<64><<<512, 256, 0, stream>>>(Yb, sum, sumsq);
  finalize_stats_kernel<<<1, 128, 0, stream>>>(sum, sumsq, g2, bt2, Ac, Bc, 64, invN);
  bnrelu_kernel<64><<<(int)(((long)NPTS * 64 / 8) / 256), 256, 0, stream>>>(Yb, Ac, Bc, Xb);

  // --- layer 3: 64 -> 128, fused BN+ReLU+maxpool epilogue ---
  wconv_kernel<<<8, 256, 0, stream>>>(W3, Wc, 64, 64, 128);
  zero_stats_kernel<<<1, 256, 0, stream>>>(sum, 256);
  gemm_wmma_kernel<64, 128><<<NPTS / 128, 256, 0, stream>>>(Wc, Xb, Yb);
  stats_kernel<128><<<512, 256, 0, stream>>>(Yb, sum, sumsq);
  finalize_stats_kernel<<<1, 128, 0, stream>>>(sum, sumsq, g3, bt3, Ac, Bc, 128, invN);
  bnmax_out_kernel<<<(B_ * S_ * 16) / 256, 256, 0, stream>>>(Yb, Ac, Bc, outFeat);
}
